// LSTMCell_44865228374504
// MI455X (gfx1250) — matrix operand order repaired and checked by reading, not compiled, not run on hardware
//
#include <hip/hip_runtime.h>
#include <hip/hip_bf16.h>

typedef __attribute__((ext_vector_type(16))) __bf16 v16bf;
typedef __attribute__((ext_vector_type(8)))  __bf16 v8bf;
typedef __attribute__((ext_vector_type(8)))  float  v8f;
typedef __attribute__((ext_vector_type(4)))  float  f32x4;

// LSTM cell, B=8192, D_IN=512, D_H=1024.  Only the g-block (cols 0..1023) and
// i-block (cols 1024..2047) of gates are needed: the reference's f/o gates
// both reuse i_in (faithful bug), halving the GEMM.  Block tile 128x64, each
// block computes the g and i column groups together so the epilogue fuses.

__device__ __forceinline__ v16bf cat8(v8bf lo, v8bf hi) {
  return __builtin_shufflevector(lo, hi, 0, 1, 2, 3, 4, 5, 6, 7,
                                 8, 9, 10, 11, 12, 13, 14, 15);
}

// Branchless fast activations: single v_exp_f32 + v_rcp_f32 (TRANS ops).
// Saturates correctly: exp2(+inf)=inf -> rcp=0; exp2(-inf)=0 -> rcp(1)=1.
__device__ __forceinline__ float fast_sigmoid(float x) {
  float e = __builtin_amdgcn_exp2f(x * -1.442695040888963f);   // exp(-x)
  return __builtin_amdgcn_rcpf(1.0f + e);
}
__device__ __forceinline__ float fast_tanh(float x) {
  float e = __builtin_amdgcn_exp2f(x * -2.885390081777927f);   // exp(-2x)
  return 1.0f - 2.0f * __builtin_amdgcn_rcpf(1.0f + e);
}

__global__ __launch_bounds__(256) void lstm_gemm_wmma(
    const float* __restrict__ x, const float* __restrict__ h,
    const float* __restrict__ c, const float* __restrict__ Wx,
    const float* __restrict__ Wh, const float* __restrict__ bias,
    float* __restrict__ outH, float* __restrict__ outC) {
  constexpr int BM = 128, BN = 64, BK = 32;
  constexpr int DIN = 512, DH = 1024, KTOT = DIN + DH;
  constexpr int NW = 4096;      // width of Wx/Wh
  constexpr int LD = BK + 8;    // pad: row stride 80B = 20 banks, conflict-free
  constexpr int NKT = KTOT / BK;

  __shared__ __align__(16) __bf16 As[BM][LD];     // [m][k]
  __shared__ __align__(16) __bf16 Bs[2][BN][LD];  // [gate][n][k] (transposed)

  const int tid  = threadIdx.x;
  const int lane = tid & 31;
  const int wv   = tid >> 5;           // wave 0..7
  const int wm   = (wv & 3) << 5;      // wave row offset 0/32/64/96
  const int wn   = (wv >> 2) << 5;     // wave col offset 0/32
  const int rowBase = blockIdx.y * BM;
  const int colBase = blockIdx.x * BN; // column in [0,1024)

  v8f accg[2][2] = {};
  v8f acci[2][2] = {};

  // Fragment addressing per ISA 7.12.2 (wave32 bf16 16x16x32).
  const int am  = lane & 15;
  const int akb = (lane >> 4) << 3;    // A: K interleave 0-7/16-23 vs 8-15/24-31
  const int bn  = lane & 15;
  const int bkb = (lane >> 4) << 3;    // B: K interleave 0-7/16-23 vs 8-15/24-31

  // ---- software-pipelined staging: global loads for step k+1 overlap the
  // ---- WMMA work of step k (single LDS buffer, regs carry the next tile).
  f32x4 aReg[4], bReg[4];

  auto stage_load = [&](int k0) {
    const bool useX = (k0 < DIN);
    const int  ks   = useX ? k0 : (k0 - DIN);
    const float* __restrict__ Ap = useX ? x : h;
    const float* __restrict__ Wp = useX ? Wx : Wh;
    const int astride = useX ? DIN : DH;
#pragma unroll
    for (int it = 0; it < 4; ++it) {
      int idx = tid + (it << 8);            // 0..1023
      int ar = idx >> 3, ac = (idx & 7) << 2;
      aReg[it] = *(const f32x4*)(Ap + (size_t)(rowBase + ar) * astride + ks + ac);
      int gate = idx >> 9, r = idx & 511;
      int kk = r >> 4, nc = (r & 15) << 2;
      bReg[it] = *(const f32x4*)(Wp + (size_t)(ks + kk) * NW +
                                 colBase + gate * DH + nc);
    }
  };
  auto stage_store = [&]() {
#pragma unroll
    for (int it = 0; it < 4; ++it) {
      int idx = tid + (it << 8);
      int ar = idx >> 3, ac = (idx & 7) << 2;
      As[ar][ac + 0] = (__bf16)aReg[it][0];
      As[ar][ac + 1] = (__bf16)aReg[it][1];
      As[ar][ac + 2] = (__bf16)aReg[it][2];
      As[ar][ac + 3] = (__bf16)aReg[it][3];
      int gate = idx >> 9, r = idx & 511;
      int kk = r >> 4, nc = (r & 15) << 2;
      Bs[gate][nc + 0][kk] = (__bf16)bReg[it][0];
      Bs[gate][nc + 1][kk] = (__bf16)bReg[it][1];
      Bs[gate][nc + 2][kk] = (__bf16)bReg[it][2];
      Bs[gate][nc + 3][kk] = (__bf16)bReg[it][3];
    }
  };

  stage_load(0);
#pragma unroll 1
  for (int kt = 0; kt < NKT; ++kt) {
    __syncthreads();
    stage_store();
    __syncthreads();
    if (kt + 1 < NKT) stage_load((kt + 1) * BK);  // overlaps compute below

    v16bf af[2];
#pragma unroll
    for (int mi = 0; mi < 2; ++mi) {
      const __bf16* p = &As[wm + mi * 16 + am][0];
      af[mi] = cat8(*(const v8bf*)(p + akb), *(const v8bf*)(p + 16 + akb));
    }
#pragma unroll
    for (int ni = 0; ni < 2; ++ni) {
#pragma unroll
      for (int g = 0; g < 2; ++g) {
        const __bf16* p = &Bs[g][wn + ni * 16 + bn][0];
        v16bf bf = cat8(*(const v8bf*)(p + bkb), *(const v8bf*)(p + 16 + bkb));
#pragma unroll
        for (int mi = 0; mi < 2; ++mi) {
          v8f& acc = g ? acci[mi][ni] : accg[mi][ni];
          acc = __builtin_amdgcn_wmma_f32_16x16x32_bf16(
              false, af[mi], false, bf, (short)0, acc, false, false);
        }
      }
    }
  }

  // Fused epilogue.  C/D layout: m = v + 8*(lane>=16), n = lane&15.
#pragma unroll
  for (int mi = 0; mi < 2; ++mi) {
#pragma unroll
    for (int ni = 0; ni < 2; ++ni) {
      const int colG = colBase + wn + ni * 16 + (lane & 15);
      const float bg = bias[colG];
      const float bi = bias[colG + DH];
#pragma unroll
      for (int vv = 0; vv < 8; ++vv) {
        const int rowG = rowBase + wm + mi * 16 + vv + ((lane >> 4) << 3);
        const size_t off = (size_t)rowG * DH + colG;
        const float gz = accg[mi][ni][vv] + bg;
        const float iz = acci[mi][ni][vv] + bi;
        const float sg = fast_sigmoid(iz);      // i_out == f_out == o_out
        const float cn = fast_tanh(gz) * sg + sg * c[off];
        outH[off] = fast_tanh(cn) * sg;
        outC[off] = cn;
      }
    }
  }
}

extern "C" void kernel_launch(void* const* d_in, const int* in_sizes, int n_in,
                              void* d_out, int out_size, void* d_ws,
                              size_t ws_size, hipStream_t stream) {
  (void)in_sizes; (void)n_in; (void)out_size; (void)d_ws; (void)ws_size;
  const float* x  = (const float*)d_in[0];
  const float* h  = (const float*)d_in[1];
  const float* c  = (const float*)d_in[2];
  const float* Wx = (const float*)d_in[3];
  const float* Wh = (const float*)d_in[4];
  const float* b  = (const float*)d_in[5];
  float* outH = (float*)d_out;
  float* outC = outH + (size_t)8192 * 1024;

  dim3 grid(1024 / 64, 8192 / 128);  // (16, 64) = 1024 workgroups
  lstm_gemm_wmma<<<grid, dim3(256), 0, stream>>>(x, h, c, Wx, Wh, b, outH, outC);
}
